// VQVAE_12790412608073
// MI455X (gfx1250) — matrix operand order repaired and checked by reading, not compiled, not run on hardware
//
#include <hip/hip_runtime.h>

// ---------------------------------------------------------------------------
// CDNA5 (gfx1250) VQ-VAE-2 forward. All convolutions run as implicit GEMM on
// v_wmma_f32_16x16x32_bf16 (bf16 inputs, f32 accumulate). wave32, WGP-coop:
// a block of up to 4 waves shares one 64-wide N tile; the 32x64 B (im2col)
// tile is staged once per K-chunk into LDS and re-read by every wave as
// 4x ds_load_b128 per WMMA operand. Weights stream in as global_load_b128.
// ---------------------------------------------------------------------------

typedef __attribute__((ext_vector_type(16))) __bf16 v16bf;
typedef __attribute__((ext_vector_type(8)))  float  v8f;

static __device__ __forceinline__ __bf16 f2bf(float f) {
  unsigned u = __builtin_bit_cast(unsigned, f);
  unsigned short h = (unsigned short)((u + 0x7FFFu + ((u >> 16) & 1u)) >> 16); // RNE
  return __builtin_bit_cast(__bf16, h);
}

// MODE 0: normal conv, kernel KH x KW (compile-time), stride/pad runtime.
//         OIHW weights == row-major M x K matrix -> vectorized A loads.
// MODE 1: ConvTranspose2d(k=4,s=2,p=1) one output-parity class (py,px);
//         each output sees exactly 2x2 taps -> K = cin*4 (IOHW gather).
// GEMM view: M = cout (grid.y x waves, 16 rows/wave), N = 16*Ho*Wo (grid.x, x64),
//            K = cin*KHKW. Spatial extents are powers of two: lgW/lgHW shifts.
template<int MODE, int KH, int KW>
__global__ void conv_wmma(
    const float* __restrict__ src1, const float* __restrict__ src2, int csplit,
    const float* __restrict__ W, const float* __restrict__ bias,
    const float* __restrict__ resid, float* __restrict__ out,
    int cin, int cout,
    int Hin, int Win, int lgHW, int lgW,
    int stride, int pad, int py, int px,
    int relu1, int relu2, int relu_out)
{
  constexpr int KHKW = (MODE == 0) ? KH * KW : 4;
  constexpr int SBS  = 36;               // padded K-stride of LDS B tile
  __shared__ float sB[64 * SBS];         // 9 KB: 32(K) x 64(N), K-major per col

  const int tid  = threadIdx.x;
  const int nthr = blockDim.x;           // 32 * nwaves
  const int wv   = tid >> 5;             // wave id in block
  const int lane = tid & 31;
  const int half = lane >> 4;            // 0: lanes 0-15, 1: lanes 16-31
  const int lid  = lane & 15;
  const int K    = cin * KHKW;
  const int m0   = (blockIdx.y * (nthr >> 5) + wv) * 16;
  const int csp2 = cin - csplit;
  const int HWmask = (1 << lgHW) - 1;
  const int Wmask  = (1 << lgW) - 1;

  // Decode the four N columns of this wave's tile (store phase).
  int nb[4], ny[4], nx[4];
#pragma unroll
  for (int t = 0; t < 4; ++t) {
    const int n = blockIdx.x * 64 + t * 16 + lid;
    nb[t] = n >> lgHW;
    const int pos = n & HWmask;
    ny[t] = pos >> lgW;
    nx[t] = pos & Wmask;
  }

  __builtin_prefetch(W, 0, 0);   // global_prefetch_b8 on the weight stream

  v8f acc[4] = {{}, {}, {}, {}};
  const int m = m0 + lid;
  const float* __restrict__ Wrow = W + (long)m * K;  // MODE 0 only

  for (int k0 = 0; k0 < K; k0 += 32) {
    // ---- Cooperative stage of B tile (32K x 64N) into LDS, once per block ----
    __syncthreads();                      // previous chunk fully consumed
    for (int i = tid; i < 32 * 64; i += nthr) {
      const int r = i >> 6;               // K row within chunk
      const int c = i & 63;               // N col within tile
      const int kIdx = k0 + r;
      float v = 0.f;
      if (kIdx < K) {
        const int n   = blockIdx.x * 64 + c;
        const int bb  = n >> lgHW;
        const int pos = n & HWmask;
        const int oy  = pos >> lgW;
        const int ox  = pos & Wmask;
        const int ci  = kIdx / KHKW;      // compile-time divisor
        const int rem = kIdx - ci * KHKW;
        int iy, ix;
        if (MODE == 0) {
          const int ky = rem / KW;
          const int kx = rem - ky * KW;
          iy = oy * stride - pad + ky;
          ix = ox * stride - pad + kx;
        } else {
          iy = oy + py - (rem >> 1);
          ix = ox + px - (rem & 1);
        }
        if (iy >= 0 && iy < Hin && ix >= 0 && ix < Win) {
          if (ci < csplit) {
            v = src1[(((long)bb * csplit + ci) * Hin + iy) * Win + ix];
            if (relu1) v = fmaxf(v, 0.f);
          } else {
            v = src2[(((long)bb * csp2 + (ci - csplit)) * Hin + iy) * Win + ix];
            if (relu2) v = fmaxf(v, 0.f);
          }
        }
      }
      sB[c * SBS + r] = v;
    }
    __syncthreads();

    // ---- A tile: weights 16(M) x 32(K) bf16, ISA per-lane K striping ----
    // Lane elements 0..7  <- K = k0+half*8 .. +7   (contiguous)
    //      elements 8..15 <- K = k0+16+half*8 .. +7 (contiguous)
    v16bf a;
    if (MODE == 0 && (k0 + 32) <= K && m < cout) {
      const int base = k0 + (half << 3);
      const float4 w0 = *(const float4*)(Wrow + base);
      const float4 w1 = *(const float4*)(Wrow + base + 4);
      const float4 w2 = *(const float4*)(Wrow + base + 16);
      const float4 w3 = *(const float4*)(Wrow + base + 20);
      a[0]  = f2bf(w0.x); a[1]  = f2bf(w0.y); a[2]  = f2bf(w0.z); a[3]  = f2bf(w0.w);
      a[4]  = f2bf(w1.x); a[5]  = f2bf(w1.y); a[6]  = f2bf(w1.z); a[7]  = f2bf(w1.w);
      a[8]  = f2bf(w2.x); a[9]  = f2bf(w2.y); a[10] = f2bf(w2.z); a[11] = f2bf(w2.w);
      a[12] = f2bf(w3.x); a[13] = f2bf(w3.y); a[14] = f2bf(w3.z); a[15] = f2bf(w3.w);
    } else {
#pragma unroll
      for (int e = 0; e < 16; ++e) {
        const int kk   = (e & 7) + ((e >> 3) << 4) + (half << 3);
        const int kIdx = k0 + kk;
        float w = 0.f;
        if (kIdx < K && m < cout) {
          if (MODE == 0) {
            w = Wrow[kIdx];                                        // OIHW row-major
          } else {
            const int ci  = kIdx >> 2;
            const int rem = kIdx & 3;
            const int ky = (1 - py) + ((rem >> 1) << 1);
            const int kx = (1 - px) + ((rem & 1) << 1);
            w = W[(((long)ci * cout + m) * 4 + ky) * 4 + kx];      // IOHW
          }
        }
        a[e] = f2bf(w);
      }
    }

    // ---- Four N sub-tiles share A; B operands read from LDS (ds_load_b128) ----
#pragma unroll
    for (int t = 0; t < 4; ++t) {
      // Lane's B elements: K rows half*8 .. +7 and 16+half*8 .. +7, fixed col t*16+lid:
      // two 8-float runs in LDS, 16B aligned -> 4x b128.
      const float* colp = &sB[(t * 16 + lid) * SBS + (half << 3)];
      const float4 b0 = *(const float4*)(colp + 0);
      const float4 b1 = *(const float4*)(colp + 4);
      const float4 b2 = *(const float4*)(colp + 16);
      const float4 b3 = *(const float4*)(colp + 20);
      v16bf bt;
      bt[0]  = f2bf(b0.x); bt[1]  = f2bf(b0.y); bt[2]  = f2bf(b0.z); bt[3]  = f2bf(b0.w);
      bt[4]  = f2bf(b1.x); bt[5]  = f2bf(b1.y); bt[6]  = f2bf(b1.z); bt[7]  = f2bf(b1.w);
      bt[8]  = f2bf(b2.x); bt[9]  = f2bf(b2.y); bt[10] = f2bf(b2.z); bt[11] = f2bf(b2.w);
      bt[12] = f2bf(b3.x); bt[13] = f2bf(b3.y); bt[14] = f2bf(b3.z); bt[15] = f2bf(b3.w);
      acc[t] = __builtin_amdgcn_wmma_f32_16x16x32_bf16(
          false, a, false, bt, (short)0, acc[t], false, false);
    }
  }

  // ---- Store: C/D layout (VGPR j: row j + 8*half, col = lane&15) ----
  const int Ho_out = (MODE == 0) ? (1 << (lgHW - lgW)) : (2 << (lgHW - lgW));
  const int Wo_out = (MODE == 0) ? (1 << lgW) : (2 << lgW);
#pragma unroll
  for (int t = 0; t < 4; ++t) {
    const int soy = (MODE == 0) ? ny[t] : (ny[t] * 2 + py);
    const int sox = (MODE == 0) ? nx[t] : (nx[t] * 2 + px);
#pragma unroll
    for (int j = 0; j < 8; ++j) {
      const int mrow = m0 + j + half * 8;
      if (mrow < cout) {
        float r = acc[t][j] + bias[mrow];
        const long idx = ((((long)nb[t] * cout + mrow) * Ho_out) + soy) * Wo_out + sox;
        if (resid) r += resid[idx];
        if (relu_out) r = fmaxf(r, 0.f);
        out[idx] = r;
      }
    }
  }
}

// Vector quantization: per-thread argmin over 512 codes (dim 64). Codebook is
// staged through 64KB LDS in two 256-code passes using CDNA5 async
// global->LDS loads (ASYNCcnt) fenced with s_wait_asynccnt.
// embed is (64, 512) row-major.
__global__ __launch_bounds__(256) void quant_kernel(
    const float* __restrict__ z, const float* __restrict__ embed,
    float* __restrict__ q, float* __restrict__ accum, int HW)
{
  __shared__ float sE[256 * 64];           // 64 KB (of 320 KB per WGP)
  const int tid = threadIdx.x;
  const int idx = blockIdx.x * 256 + tid;  // grid sized exactly
  const int b   = idx / HW;
  const int pos = idx - b * HW;
  const float* zp = z + ((long)b * 64) * HW + pos;

  float zf[64];
#pragma unroll
  for (int d = 0; d < 64; ++d) zf[d] = zp[(long)d * HW];

  float best = 3.4e38f;
  int   bi   = 0;
  for (int p = 0; p < 2; ++p) {
    __syncthreads();
    // Async direct-to-LDS stage of 256 codes (transposed to code-major).
    for (int i = tid; i < 256 * 64; i += 256) {
      const int d = i >> 8;
      const int e = i & 255;
      const float* gp = embed + d * 512 + p * 256 + e;
      const unsigned loff = (unsigned)(unsigned long long)(const void*)&sE[e * 64 + d];
      asm volatile("global_load_async_to_lds_b32 %0, %1, off"
                   :: "v"(loff), "v"(gp) : "memory");
    }
    asm volatile("s_wait_asynccnt 0" ::: "memory");
    __syncthreads();
    for (int e = 0; e < 256; ++e) {
      const float* ev = sE + e * 64;
      float dot = 0.f, nn = 0.f;
#pragma unroll
      for (int d = 0; d < 64; ++d) {
        const float v = ev[d];
        dot += zf[d] * v;
        nn  += v * v;
      }
      const float cost = nn - 2.f * dot;   // ||z||^2 constant per row
      if (cost < best) { best = cost; bi = p * 256 + e; }
    }
  }

  float* qp = q + ((long)b * 64) * HW + pos;
  float local = 0.f;
#pragma unroll
  for (int d = 0; d < 64; ++d) {
    const float qd = embed[d * 512 + bi];
    const float df = qd - zf[d];
    local += df * df;
    qp[(long)d * HW] = qd;
  }
  // wave32 tree reduction, one atomic per wave
#pragma unroll
  for (int s = 16; s > 0; s >>= 1) local += __shfl_xor(local, s, 32);
  if ((tid & 31) == 0) atomicAdd(accum, local);
}

__global__ void zero2_kernel(float* p) {
  if (threadIdx.x < 2) p[threadIdx.x] = 0.f;
}

__global__ void finalize_kernel(const float* __restrict__ sums, float* __restrict__ od) {
  if (threadIdx.x == 0)
    od[0] = sums[0] * (1.0f / 1048576.0f) + sums[1] * (1.0f / 4194304.0f);
}

// ---------------------------------------------------------------------------
// Host-side launch helpers
// ---------------------------------------------------------------------------
static inline void cv(hipStream_t st, const float* s1, const float* s2, int split,
                      const float* W, const float* bias, const float* resid, float* out,
                      int cin, int cout, int Hin, int Win, int Ho, int Wo,
                      int k, int sd, int pad, int r1, int r2, int ro)
{
  const int N = 16 * Ho * Wo;
  const int mtiles = (cout + 15) / 16;
  const int nw = mtiles < 4 ? mtiles : 4;
  dim3 g(N / 64, (mtiles + nw - 1) / nw);
  dim3 bdim(32 * nw);
  const int lgW  = __builtin_ctz((unsigned)Wo);
  const int lgHW = __builtin_ctz((unsigned)(Ho * Wo));
  const float* s2p = s2 ? s2 : s1;
  if (k == 4)
    conv_wmma<0, 4, 4><<<g, bdim, 0, st>>>(s1, s2p, split, W, bias, resid, out,
        cin, cout, Hin, Win, lgHW, lgW, sd, pad, 0, 0, r1, r2, ro);
  else if (k == 3)
    conv_wmma<0, 3, 3><<<g, bdim, 0, st>>>(s1, s2p, split, W, bias, resid, out,
        cin, cout, Hin, Win, lgHW, lgW, sd, pad, 0, 0, r1, r2, ro);
  else
    conv_wmma<0, 1, 1><<<g, bdim, 0, st>>>(s1, s2p, split, W, bias, resid, out,
        cin, cout, Hin, Win, lgHW, lgW, sd, pad, 0, 0, r1, r2, ro);
}

static inline void cvT(hipStream_t st, const float* s1, const float* W, const float* bias,
                       float* out, int cin, int cout, int Hin, int Win, int r1, int ro)
{
  const int N = 16 * Hin * Win;
  const int mtiles = (cout + 15) / 16;
  const int nw = mtiles < 4 ? mtiles : 4;
  dim3 g(N / 64, (mtiles + nw - 1) / nw);
  dim3 bdim(32 * nw);
  const int lgW  = __builtin_ctz((unsigned)Win);
  const int lgHW = __builtin_ctz((unsigned)(Hin * Win));
  for (int py = 0; py < 2; ++py)
    for (int px = 0; px < 2; ++px)
      conv_wmma<1, 4, 4><<<g, bdim, 0, st>>>(s1, s1, cin, W, bias, nullptr, out,
          cin, cout, Hin, Win, lgHW, lgW, 2, 1, py, px, r1, 0, ro);
}

static inline void resblock(hipStream_t st, float* X, float* T,
                            const float* rw1, const float* rb1,
                            const float* rw2, const float* rb2, int Hs)
{
  for (int i = 0; i < 2; ++i) {
    // t = conv3x3(relu(X)) : 128 -> 32
    cv(st, X, nullptr, 128, rw1 + (long)i * 32 * 128 * 9, rb1 + i * 32, nullptr, T,
       128, 32, Hs, Hs, Hs, Hs, 3, 1, 1, 1, 0, 0);
    // X += conv1x1(relu(t)) : 32 -> 128  (in-place safe: 1x1, per-element)
    cv(st, T, nullptr, 32, rw2 + (long)i * 128 * 32, rb2 + i * 128, X, X,
       32, 128, Hs, Hs, Hs, Hs, 1, 1, 0, 1, 0, 0);
  }
}

extern "C" void kernel_launch(void* const* d_in, const int* in_sizes, int n_in,
                              void* d_out, int out_size, void* d_ws, size_t ws_size,
                              hipStream_t stream)
{
  (void)in_sizes; (void)n_in; (void)out_size; (void)ws_size;

  const float* x       = (const float*)d_in[0];
  const float* eb_w0   = (const float*)d_in[1];
  const float* eb_b0   = (const float*)d_in[2];
  const float* eb_w1   = (const float*)d_in[3];
  const float* eb_b1   = (const float*)d_in[4];
  const float* eb_w2   = (const float*)d_in[5];
  const float* eb_b2   = (const float*)d_in[6];
  const float* eb_rw1  = (const float*)d_in[7];
  const float* eb_rb1  = (const float*)d_in[8];
  const float* eb_rw2  = (const float*)d_in[9];
  const float* eb_rb2  = (const float*)d_in[10];
  const float* et_w0   = (const float*)d_in[11];
  const float* et_b0   = (const float*)d_in[12];
  const float* et_w1   = (const float*)d_in[13];
  const float* et_b1   = (const float*)d_in[14];
  const float* et_rw1  = (const float*)d_in[15];
  const float* et_rb1  = (const float*)d_in[16];
  const float* et_rw2  = (const float*)d_in[17];
  const float* et_rb2  = (const float*)d_in[18];
  const float* qt_w    = (const float*)d_in[19];
  const float* qt_b    = (const float*)d_in[20];
  const float* embed_t = (const float*)d_in[21];
  const float* dt_w0   = (const float*)d_in[22];
  const float* dt_b0   = (const float*)d_in[23];
  const float* dt_rw1  = (const float*)d_in[24];
  const float* dt_rb1  = (const float*)d_in[25];
  const float* dt_rw2  = (const float*)d_in[26];
  const float* dt_rb2  = (const float*)d_in[27];
  const float* dt_wT   = (const float*)d_in[28];
  const float* dt_bT   = (const float*)d_in[29];
  const float* qb_w    = (const float*)d_in[30];
  const float* qb_b    = (const float*)d_in[31];
  const float* embed_b = (const float*)d_in[32];
  const float* up_wT   = (const float*)d_in[33];
  const float* up_bT   = (const float*)d_in[34];
  const float* dw0     = (const float*)d_in[35];
  const float* db0     = (const float*)d_in[36];
  const float* d_rw1   = (const float*)d_in[37];
  const float* d_rb1   = (const float*)d_in[38];
  const float* d_rw2   = (const float*)d_in[39];
  const float* d_rb2   = (const float*)d_in[40];
  const float* d_wT1   = (const float*)d_in[41];
  const float* d_bT1   = (const float*)d_in[42];
  const float* d_wT2   = (const float*)d_in[43];
  const float* d_bT2   = (const float*)d_in[44];

  float* ws   = (float*)d_ws;
  float* H0   = ws + 0;          // 16*64*128*128  (reused later as D1)
  float* H1   = ws + 16777216;   // 16*128*64*64
  float* H2   = ws + 25165824;   // 16*128*64*64  (enc_b pre-relu; alive -> zb)
  float* T    = ws + 33554432;   // res temp (max 16*32*64*64)
  float* ET0  = ws + 35651584;   // 16*64*32*32
  float* ET1  = ws + 36700160;   // 16*128*32*32  (enc_t pre-relu)
  float* ZT   = ws + 38797312;   // 16*64*32*32
  float* QT   = ws + 39845888;   // 16*64*32*32   (alive -> up_t)
  float* DT0  = ws + 40894464;   // 16*128*32*32
  float* DECT = ws + 42991616;   // 16*64*64*64
  float* ZB   = ws + 47185920;   // 16*64*64*64
  float* QB   = ws + 51380224;   // 16*64*64*64
  float* UPT  = ws + 55574528;   // 16*64*64*64
  float* D0   = ws + 59768832;   // 16*128*64*64
  float* SUM  = ws + 68157440;   // 2 floats (diff_t, diff_b sums)
  float* D1   = H0;              // 16*64*128*128 (H0 dead by then)

  float* dec = (float*)d_out;    // 16*3*256*256 floats, then 1 diff float

  // ---- Encoder bottom (256 -> 64) ----
  cv(stream, x,  nullptr, 3,   eb_w0, eb_b0, nullptr, H0, 3,  64, 256,256, 128,128, 4,2,1, 0,0,1);
  cv(stream, H0, nullptr, 64,  eb_w1, eb_b1, nullptr, H1, 64, 128, 128,128, 64,64,  4,2,1, 0,0,1);
  cv(stream, H1, nullptr, 128, eb_w2, eb_b2, nullptr, H2, 128,128, 64,64,  64,64,  3,1,1, 0,0,0);
  resblock(stream, H2, T, eb_rw1, eb_rb1, eb_rw2, eb_rb2, 64);
  // enc_b = relu(H2) applied lazily via relu_in at consumers.

  // ---- Encoder top (64 -> 32) ----
  cv(stream, H2,  nullptr, 128, et_w0, et_b0, nullptr, ET0, 128,64, 64,64, 32,32, 4,2,1, 1,0,1);
  cv(stream, ET0, nullptr, 64,  et_w1, et_b1, nullptr, ET1, 64,128, 32,32, 32,32, 3,1,1, 0,0,0);
  resblock(stream, ET1, T, et_rw1, et_rb1, et_rw2, et_rb2, 32);

  // ---- Quantize top ----
  cv(stream, ET1, nullptr, 128, qt_w, qt_b, nullptr, ZT, 128,64, 32,32, 32,32, 1,1,0, 1,0,0);
  zero2_kernel<<<1, 32, 0, stream>>>(SUM);
  quant_kernel<<<(16 * 1024) / 256, 256, 0, stream>>>(ZT, embed_t, QT, SUM + 0, 1024);

  // ---- Decode top -> bottom resolution ----
  cv(stream, QT, nullptr, 64, dt_w0, dt_b0, nullptr, DT0, 64,128, 32,32, 32,32, 3,1,1, 0,0,0);
  resblock(stream, DT0, T, dt_rw1, dt_rb1, dt_rw2, dt_rb2, 32);
  cvT(stream, DT0, dt_wT, dt_bT, DECT, 128, 64, 32, 32, 1, 0);   // relu(res) -> convT

  // ---- Quantize bottom (concat(dec_t, relu(enc_b)) fused via dual source) ----
  cv(stream, DECT, H2, 64, qb_w, qb_b, nullptr, ZB, 192,64, 64,64, 64,64, 1,1,0, 0,1,0);
  quant_kernel<<<(16 * 4096) / 256, 256, 0, stream>>>(ZB, embed_b, QB, SUM + 1, 4096);

  // ---- Decode to image ----
  cvT(stream, QT, up_wT, up_bT, UPT, 64, 64, 32, 32, 0, 0);
  cv(stream, UPT, QB, 64, dw0, db0, nullptr, D0, 128,128, 64,64, 64,64, 3,1,1, 0,0,0);
  resblock(stream, D0, T, d_rw1, d_rb1, d_rw2, d_rb2, 64);
  cvT(stream, D0, d_wT1, d_bT1, D1, 128, 64, 64, 64, 1, 1);      // relu in, relu out
  cvT(stream, D1, d_wT2, d_bT2, dec, 64, 3, 128, 128, 0, 0);

  finalize_kernel<<<1, 32, 0, stream>>>(SUM, dec + (long)16 * 3 * 256 * 256);
}
